// ScaledDotProductAttention_3049426780750
// MI455X (gfx1250) — matrix operand order repaired and checked by reading, not compiled, not run on hardware
//
#include <hip/hip_runtime.h>

typedef _Float16 hv2  __attribute__((ext_vector_type(2)));
typedef _Float16 hv4  __attribute__((ext_vector_type(4)));
typedef _Float16 hv8  __attribute__((ext_vector_type(8)));
typedef _Float16 v16h __attribute__((ext_vector_type(16)));
typedef float    v8f  __attribute__((ext_vector_type(8)));

namespace {
constexpr int   kB = 32;
constexpr int   kS = 2048;
constexpr int   kD = 128;
// scale * log2(e): scores come out of QK^T ready for exp2
constexpr float kQScale = 0.08838834764831845f * 1.4426950408889634f;

constexpr int QROWS = 128;  // query rows per workgroup (8 waves x 16 rows)
constexpr int KT    = 64;   // key/value rows per tile
constexpr int KSTR  = 136;  // K LDS row stride (halves), 272B = 16*17
constexpr int VTS   = 72;   // V^T row stride, 144B = 16*9
constexpr int PS    = 72;   // P row stride, 144B
}

__device__ __forceinline__ v8f wmma_f16(v16h a, v16h b, v8f c) {
  return __builtin_amdgcn_wmma_f32_16x16x32_f16(false, a, false, b, (short)0, c,
                                                false, false);
}

// 16-bit A fragment (16x32): lanes 0-15 row M=lane hold K = kk..kk+15;
// lanes 16-31 same row hold K = kk+16..kk+31.
__device__ __forceinline__ v16h load_a_frag(const _Float16* row, int kk, int sel) {
  v16h a;
  hv8* ap = reinterpret_cast<hv8*>(&a);
  ap[0] = *reinterpret_cast<const hv8*>(row + kk + sel * 16);
  ap[1] = *reinterpret_cast<const hv8*>(row + kk + sel * 16 + 8);
  return a;
}

// 16-bit B fragment (32x16): lane holds 16 contiguous K values of one column.
__device__ __forceinline__ v16h load_b_frag(const _Float16* p) {
  v16h b;
  hv8* bp = reinterpret_cast<hv8*>(&b);
  bp[0] = *reinterpret_cast<const hv8*>(p);
  bp[1] = *reinterpret_cast<const hv8*>(p + 8);
  return b;
}

template <int MASK>
__device__ __forceinline__ float swz_xor(float x) {
  int y = __builtin_amdgcn_ds_swizzle(__builtin_bit_cast(int, x),
                                      (MASK << 10) | 0x1f);
  return __builtin_bit_cast(float, y);
}

// one batched reduction round: 8 swizzles, one wait, 8 maxes
template <int STEP>
__device__ __forceinline__ void max_round(float (&mx)[8]) {
  float t[8];
#pragma unroll
  for (int r = 0; r < 8; ++r) t[r] = swz_xor<STEP>(mx[r]);
#pragma unroll
  for (int r = 0; r < 8; ++r) mx[r] = fmaxf(mx[r], t[r]);
}

__global__ __launch_bounds__(256, 1)
void fa_fwd_kernel(const float* __restrict__ q, const float* __restrict__ k,
                   const float* __restrict__ v, float* __restrict__ out) {
  __shared__ __align__(16) _Float16 Ksh[KT][KSTR];   // 17408 B
  __shared__ __align__(16) _Float16 Vt [kD][VTS];    // 18432 B  (V^T: [d][key])
  __shared__ __align__(16) _Float16 Psh[8][16][PS];  // 18432 B

  const int tid  = threadIdx.x;
  const int w    = tid >> 5;    // wave 0..7
  const int lane = tid & 31;
  const int ln   = lane & 15;
  const int sel  = lane >> 4;

  const int qbase   = blockIdx.x * QROWS;
  const int batch   = blockIdx.y;
  const size_t boff = (size_t)batch * kS * kD;

  // ---- load wave's Q A-fragments straight from global (one time) ----
  v16h qa[4];
  {
    const float* qrow = q + boff + (size_t)(qbase + w * 16 + ln) * kD;
#pragma unroll
    for (int c = 0; c < 4; ++c) {
      const int base = c * 32 + sel * 16;
      const float4 f0 = *reinterpret_cast<const float4*>(qrow + base);
      const float4 f1 = *reinterpret_cast<const float4*>(qrow + base + 4);
      const float4 f2 = *reinterpret_cast<const float4*>(qrow + base + 8);
      const float4 f3 = *reinterpret_cast<const float4*>(qrow + base + 12);
      union { v16h v; hv2 p[8]; } u;
      u.p[0] = hv2{(_Float16)(f0.x * kQScale), (_Float16)(f0.y * kQScale)};
      u.p[1] = hv2{(_Float16)(f0.z * kQScale), (_Float16)(f0.w * kQScale)};
      u.p[2] = hv2{(_Float16)(f1.x * kQScale), (_Float16)(f1.y * kQScale)};
      u.p[3] = hv2{(_Float16)(f1.z * kQScale), (_Float16)(f1.w * kQScale)};
      u.p[4] = hv2{(_Float16)(f2.x * kQScale), (_Float16)(f2.y * kQScale)};
      u.p[5] = hv2{(_Float16)(f2.z * kQScale), (_Float16)(f2.w * kQScale)};
      u.p[6] = hv2{(_Float16)(f3.x * kQScale), (_Float16)(f3.y * kQScale)};
      u.p[7] = hv2{(_Float16)(f3.z * kQScale), (_Float16)(f3.w * kQScale)};
      qa[c] = u.v;
    }
  }

  // all-ones B fragment: l_acc += P @ ones gives row sums of P
  v16h ones;
#pragma unroll
  for (int i = 0; i < 16; ++i) ones[i] = (_Float16)1.0f;

  v8f zero = {};
  v8f o[8];
#pragma unroll
  for (int t = 0; t < 8; ++t) o[t] = zero;
  v8f l_acc = zero;
  float m_run[8];
#pragma unroll
  for (int r = 0; r < 8; ++r) m_run[r] = -3.0e38f;

  for (int jn = 0; jn < kS; jn += KT) {
    __syncthreads();  // previous K/V tile fully consumed

    // ---- stage K tile (row-major f16) ----
    {
      const int krow = tid >> 2;  // 0..63
      const float4* kr4 =
          reinterpret_cast<const float4*>(k + boff + (size_t)(jn + krow) * kD);
#pragma unroll
      for (int j = 0; j < 8; ++j) {
        const int c4 = (tid & 3) * 8 + j;
        const float4 kv = kr4[c4];
        hv4 hk = { (_Float16)kv.x, (_Float16)kv.y,
                   (_Float16)kv.z, (_Float16)kv.w };
        *reinterpret_cast<hv4*>(&Ksh[krow][c4 * 4]) = hk;
      }
      if (jn + KT < kS) __builtin_prefetch(kr4 + KT * kD / 4, 0, 1);
    }
    // ---- stage V tile transposed: Vt[d][key] ----
    {
      const int d0 = (tid & 31) * 4;
      const int kg = (tid >> 5) * 8;  // 8 waves x 8 keys
      hv8 vbuf[4];
#pragma unroll
      for (int j = 0; j < 8; ++j) {
        const float4 vv = *reinterpret_cast<const float4*>(
            v + boff + (size_t)(jn + kg + j) * kD + d0);
        vbuf[0][j] = (_Float16)vv.x;
        vbuf[1][j] = (_Float16)vv.y;
        vbuf[2][j] = (_Float16)vv.z;
        vbuf[3][j] = (_Float16)vv.w;
      }
#pragma unroll
      for (int i = 0; i < 4; ++i)
        *reinterpret_cast<hv8*>(&Vt[d0 + i][kg]) = vbuf[i];
      if (jn + KT < kS)
        __builtin_prefetch(v + boff + (size_t)(jn + KT + kg) * kD + d0, 0, 1);
    }
    __syncthreads();

    // ---- S = Q K^T : 16 x 64 block per wave, software-pipelined B loads ----
    v8f s[4];
#pragma unroll
    for (int i = 0; i < 4; ++i) s[i] = zero;
    {
      v16h bcur = load_b_frag(&Ksh[ln][sel * 16]);  // (c=0, i=0)
#pragma unroll
      for (int c = 0; c < 4; ++c) {
#pragma unroll
        for (int i = 0; i < 4; ++i) {
          v16h bnext = bcur;
          if (!(c == 3 && i == 3)) {
            const int ni = (i + 1) & 3;
            const int nc = (i == 3) ? c + 1 : c;
            bnext = load_b_frag(&Ksh[ni * 16 + ln][nc * 32 + sel * 16]);
          }
          s[i] = wmma_f16(qa[c], bcur, s[i]);
          bcur = bnext;
        }
      }
    }

    // ---- online softmax (base-2 domain; scores pre-scaled) ----
    float mx[8];
#pragma unroll
    for (int r = 0; r < 8; ++r)
      mx[r] = fmaxf(fmaxf(s[0][r], s[1][r]), fmaxf(s[2][r], s[3][r]));
    max_round<1>(mx);
    max_round<2>(mx);
    max_round<4>(mx);
    max_round<8>(mx);

    v8f alv;
#pragma unroll
    for (int r = 0; r < 8; ++r) {
      const float mn = fmaxf(m_run[r], mx[r]);
      alv[r] = __builtin_amdgcn_exp2f(m_run[r] - mn);
      m_run[r] = mn;
    }
    // p = 2^(s - m), spill to LDS in C-layout for A-layout regather
#pragma unroll
    for (int r = 0; r < 8; ++r) {
      const float mn = m_run[r];
#pragma unroll
      for (int i = 0; i < 4; ++i) {
        const float e = __builtin_amdgcn_exp2f(s[i][r] - mn);
        Psh[w][sel * 8 + r][i * 16 + ln] = (_Float16)e;
      }
    }
    // rescale accumulators only when the running max actually moved
    float amin = fminf(fminf(fminf(alv[0], alv[1]), fminf(alv[2], alv[3])),
                       fminf(fminf(alv[4], alv[5]), fminf(alv[6], alv[7])));
    if (__ballot(amin < 1.0f)) {
      l_acc *= alv;
#pragma unroll
      for (int t = 0; t < 8; ++t) o[t] *= alv;
    }

    asm volatile("s_wait_dscnt 0" ::: "memory");  // P stores visible

    // ---- O += P(16x64) @ V(64x128); l += P @ ones (pipelined) ----
    const v16h pa0 = load_a_frag(&Psh[w][ln][0], 0, sel);
    const v16h pa1 = load_a_frag(&Psh[w][ln][0], 32, sel);
    v16h vb0 = load_b_frag(&Vt[ln][sel * 16]);        // t = 0 fragments
    v16h vb1 = load_b_frag(&Vt[ln][32 + sel * 16]);
    // these only need pa0/pa1; V loads stay in flight underneath
    l_acc = wmma_f16(pa0, ones, l_acc);
    l_acc = wmma_f16(pa1, ones, l_acc);
#pragma unroll
    for (int t = 0; t < 8; ++t) {
      v16h nb0 = vb0, nb1 = vb1;
      if (t < 7) {
        nb0 = load_b_frag(&Vt[(t + 1) * 16 + ln][sel * 16]);
        nb1 = load_b_frag(&Vt[(t + 1) * 16 + ln][32 + sel * 16]);
      }
      o[t] = wmma_f16(pa0, vb0, o[t]);
      o[t] = wmma_f16(pa1, vb1, o[t]);
      vb0 = nb0;
      vb1 = nb1;
    }
  }

  // ---- normalize and write O (fp32) ----
#pragma unroll
  for (int r = 0; r < 8; ++r) {
    const float invl = 1.0f / l_acc[r];
    float* po = out +
        ((size_t)batch * kS + qbase + w * 16 + sel * 8 + r) * kD + ln;
#pragma unroll
    for (int t = 0; t < 8; ++t) po[t * 16] = o[t][r] * invl;
  }
}

extern "C" void kernel_launch(void* const* d_in, const int* in_sizes, int n_in,
                              void* d_out, int out_size, void* d_ws,
                              size_t ws_size, hipStream_t stream) {
  (void)in_sizes; (void)n_in; (void)out_size; (void)d_ws; (void)ws_size;
  const float* q = (const float*)d_in[0];
  const float* k = (const float*)d_in[1];
  const float* v = (const float*)d_in[2];
  float* out = (float*)d_out;
  dim3 grid(kS / QROWS, kB);
  fa_fwd_kernel<<<grid, 256, 0, stream>>>(q, k, v, out);
}
